// RBF_16466904613581
// MI455X (gfx1250) — matrix operand order repaired and checked by reading, not compiled, not run on hardware
//
#include <hip/hip_runtime.h>
#include <hip/hip_bf16.h>

#define N_INPUT 16384
#define NUM_C   4096
#define DIM     256
#define KTOT    512          // concatenated K: [x^2 | x] vs [inv2s2 | -2*centers*inv2s2]

#define TILE_M  128
#define TILE_N  128
#define TILE_K  64
#define LDKB    72           // bf16 row stride: 144 B (16B-aligned), 36-word bank spread
#define NKTILES (KTOT / TILE_K)

typedef float  v8f   __attribute__((ext_vector_type(8)));
typedef __bf16 v8bf  __attribute__((ext_vector_type(8)));
typedef __bf16 v16bf __attribute__((ext_vector_type(16)));

// ---------------- prep: split A_cat = [x^2 | x] into bf16 hi/lo planes
__global__ __launch_bounds__(256) void rbf_prep_a(
    const float* __restrict__ x, __bf16* __restrict__ AH, __bf16* __restrict__ AL) {
  int idx = blockIdx.x * 256 + threadIdx.x;  // 0 .. N*D-1
  int n = idx >> 8;
  int d = idx & 255;
  float v = x[idx];
  float sq = v * v;
  __bf16 h1 = (__bf16)sq;
  AH[(size_t)n * KTOT + d] = h1;
  AL[(size_t)n * KTOT + d] = (__bf16)(sq - (float)h1);
  __bf16 h2 = (__bf16)v;
  AH[(size_t)n * KTOT + DIM + d] = h2;
  AL[(size_t)n * KTOT + DIM + d] = (__bf16)(v - (float)h2);
}

// ---------------- prep: split B_cat = [inv2s2 | -2*centers*inv2s2] into bf16 hi/lo planes
__global__ __launch_bounds__(256) void rbf_prep_b(
    const float* __restrict__ centers, const float* __restrict__ sigmas,
    __bf16* __restrict__ BH, __bf16* __restrict__ BL) {
  int idx = blockIdx.x * 256 + threadIdx.x;  // 0 .. C*D-1
  int c = idx >> 8;
  int d = idx & 255;
  float s = sigmas[idx];
  float inv = 1.0f / (2.0f * s * s);
  float ce = centers[idx];
  __bf16 h1 = (__bf16)inv;
  BH[(size_t)c * KTOT + d] = h1;
  BL[(size_t)c * KTOT + d] = (__bf16)(inv - (float)h1);
  float cr = -2.0f * ce * inv;
  __bf16 h2 = (__bf16)cr;
  BH[(size_t)c * KTOT + DIM + d] = h2;
  BL[(size_t)c * KTOT + DIM + d] = (__bf16)(cr - (float)h2);
}

// ---------------- prep: c2[c] = sum_d centers^2 * inv2s2 (kept in f32)
__global__ __launch_bounds__(256) void rbf_prep_c2(
    const float* __restrict__ centers, const float* __restrict__ sigmas,
    float* __restrict__ c2) {
  __shared__ float red[256];
  int c = blockIdx.x;
  int d = threadIdx.x;
  float s = sigmas[c * DIM + d];
  float inv = 1.0f / (2.0f * s * s);
  float ce = centers[c * DIM + d];
  red[d] = ce * ce * inv;
  __syncthreads();
  for (int off = 128; off > 0; off >>= 1) {
    if (d < off) red[d] += red[d + off];
    __syncthreads();
  }
  if (d == 0) c2[c] = red[0];
}

// async 16-byte global -> LDS copy (ASYNCcnt-tracked, no VGPR data path)
static __device__ __forceinline__ void async_copy16(unsigned lds_off, const void* gptr) {
  asm volatile("global_load_async_to_lds_b128 %0, %1, off"
               :: "v"(lds_off), "v"(gptr)
               : "memory");
}
static __device__ __forceinline__ void wait_async16() {
  asm volatile("s_wait_asynccnt 0x10" ::: "memory");  // <=16 outstanding (the prefetch)
}
static __device__ __forceinline__ void wait_async0() {
  asm volatile("s_wait_asynccnt 0x0" ::: "memory");
}

// concat two 8-element bf16 LDS loads into a 16-element WMMA fragment
static __device__ __forceinline__ v16bf frag16(const __bf16* p0, const __bf16* p1) {
  v8bf a = *(const v8bf*)p0;
  v8bf b = *(const v8bf*)p1;
  return __builtin_shufflevector(a, b, 0, 1, 2, 3, 4, 5, 6, 7,
                                       8, 9, 10, 11, 12, 13, 14, 15);
}

// ---------------- main fused GEMM (bf16x3 compensated split) + exp + weighted row-reduction
// grid = (N/128, C/128), block = 256 (8 waves); wave grid 4(M) x 2(N),
// each wave computes a 32x64 region = 2x4 WMMA 16x16 tiles.
// Double-buffered LDS, filled by async global->LDS copies of pre-split bf16 planes.
__global__ __launch_bounds__(256) void rbf_main(
    const __bf16* __restrict__ AH, const __bf16* __restrict__ AL,
    const __bf16* __restrict__ BH, const __bf16* __restrict__ BL,
    const float* __restrict__ c2, const float* __restrict__ w,
    float* __restrict__ partials) {
  __shared__ __bf16 AsH[2][TILE_M][LDKB];
  __shared__ __bf16 AsL[2][TILE_M][LDKB];
  __shared__ __bf16 BsH[2][TILE_N][LDKB];
  __shared__ __bf16 BsL[2][TILE_N][LDKB];
  __shared__ float rowsum[TILE_M];

  const int tid  = threadIdx.x;
  const int lane = tid & 31;
  const int wv   = tid >> 5;
  const int mBase = (wv >> 1) * 32;  // wave row base within tile
  const int nBase = (wv & 1) * 64;   // wave col base within tile

  const int nb = blockIdx.x * TILE_M;  // global N base
  const int cb = blockIdx.y * TILE_N;  // global C base

  if (tid < TILE_M) rowsum[tid] = 0.0f;

  v8f acc[2][4];
  for (int mi = 0; mi < 2; ++mi)
    for (int ni = 0; ni < 4; ++ni)
      for (int e = 0; e < 8; ++e) acc[mi][ni][e] = 0.0f;

  const int m16  = lane & 15;
  const int aklo = (lane >> 4) << 3;  // A frag: lane half picks K sub-blocks {0-7,16-23} / {8-15,24-31}
  const int bklo = (lane >> 4) << 3;  // B frag: lane half picks K sub-blocks {0-7,16-23} / {8-15,24-31}

  // stage one K-tile (TILE_K=64 bf16 = 128 B/row) into LDS buffer `buf` via async copies
  auto stage = [&](int buf, int kb) {
    #pragma unroll
    for (int t = 0; t < 4; ++t) {
      int idx = tid + t * 256;     // 0..1023 16-byte chunks per plane
      int row = idx >> 3;          // 8 chunks per 128-byte row
      int ch  = (idx & 7) << 3;    // chunk offset in bf16 elements (8 per 16B)
      size_t gaOff = ((size_t)(nb + row) * KTOT + kb + ch);
      size_t gbOff = ((size_t)(cb + row) * KTOT + kb + ch);
      async_copy16((unsigned)(uintptr_t)&AsH[buf][row][ch], AH + gaOff);
      async_copy16((unsigned)(uintptr_t)&AsL[buf][row][ch], AL + gaOff);
      async_copy16((unsigned)(uintptr_t)&BsH[buf][row][ch], BH + gbOff);
      async_copy16((unsigned)(uintptr_t)&BsL[buf][row][ch], BL + gbOff);
    }
  };

  stage(0, 0);  // prologue prefetch

  for (int ki = 0; ki < NKTILES; ++ki) {
    const int buf = ki & 1;
    if (ki + 1 < NKTILES) {
      stage(buf ^ 1, (ki + 1) * TILE_K);  // prefetch next tile (prev compute barrier makes this safe)
      wait_async16();                      // current tile's 16 copies done; prefetch still in flight
    } else {
      wait_async0();
    }
    __syncthreads();  // all waves' copies for `buf` complete

    #pragma unroll
    for (int kk = 0; kk < TILE_K; kk += 32) {
      v16bf aH[2], aL[2];
      #pragma unroll
      for (int mi = 0; mi < 2; ++mi) {
        const __bf16* pH = &AsH[buf][mBase + mi * 16 + m16][kk + aklo];
        const __bf16* pL = &AsL[buf][mBase + mi * 16 + m16][kk + aklo];
        aH[mi] = frag16(pH, pH + 16);
        aL[mi] = frag16(pL, pL + 16);
      }
      #pragma unroll
      for (int ni = 0; ni < 4; ++ni) {
        const __bf16* pH = &BsH[buf][nBase + ni * 16 + m16][kk + bklo];
        const __bf16* pL = &BsL[buf][nBase + ni * 16 + m16][kk + bklo];
        v16bf bHf = frag16(pH, pH + 16);
        v16bf bLf = frag16(pL, pL + 16);
        #pragma unroll
        for (int mi = 0; mi < 2; ++mi) {
          // bf16x3 compensated product: hi*hi + hi*lo + lo*hi (f32 accumulate)
          acc[mi][ni] = __builtin_amdgcn_wmma_f32_16x16x32_bf16(
              false, aH[mi], false, bHf, (short)0, acc[mi][ni], false, false);
          acc[mi][ni] = __builtin_amdgcn_wmma_f32_16x16x32_bf16(
              false, aH[mi], false, bLf, (short)0, acc[mi][ni], false, false);
          acc[mi][ni] = __builtin_amdgcn_wmma_f32_16x16x32_bf16(
              false, aL[mi], false, bHf, (short)0, acc[mi][ni], false, false);
        }
      }
    }
    __syncthreads();  // all waves done reading `buf` before it is overwritten
  }

  // ---- epilogue: val = exp(-(acc + c2[c])) * w[c]; reduce over this tile's C cols
  float c2v[4], wv4[4];
  #pragma unroll
  for (int ni = 0; ni < 4; ++ni) {
    int c = cb + nBase + ni * 16 + m16;
    c2v[ni] = c2[c];
    wv4[ni] = w[c];
  }
  const int rowOff8 = (lane >> 4) << 3;  // lanes 16-31 hold rows M=8..15 of each C/D tile
  #pragma unroll
  for (int mi = 0; mi < 2; ++mi) {
    #pragma unroll
    for (int r = 0; r < 8; ++r) {
      float sum = 0.0f;
      #pragma unroll
      for (int ni = 0; ni < 4; ++ni) {
        float d2 = acc[mi][ni][r] + c2v[ni];
        sum += __expf(-d2) * wv4[ni];
      }
      // reduce across the 16 lanes that hold the same matrix row
      sum += __shfl_xor(sum, 1, 32);
      sum += __shfl_xor(sum, 2, 32);
      sum += __shfl_xor(sum, 4, 32);
      sum += __shfl_xor(sum, 8, 32);
      if (m16 == 0)  // one lane per 16-lane group; 2 waves/row -> commutative adds
        atomicAdd(&rowsum[mBase + mi * 16 + rowOff8 + r], sum);
    }
  }
  __syncthreads();
  if (tid < TILE_M)
    partials[(size_t)(nb + tid) * (NUM_C / TILE_N) + blockIdx.y] = rowsum[tid];
}

// ---------------- final: reduce C-block partials, bias, sigmoid
__global__ __launch_bounds__(256) void rbf_finish(
    const float* __restrict__ partials, const float* __restrict__ b_lin,
    float* __restrict__ out) {
  int n = blockIdx.x * 256 + threadIdx.x;
  float s = b_lin[0];
  #pragma unroll
  for (int j = 0; j < NUM_C / TILE_N; ++j)
    s += partials[(size_t)n * (NUM_C / TILE_N) + j];
  out[n] = 1.0f / (1.0f + __expf(-s));
}

extern "C" void kernel_launch(void* const* d_in, const int* in_sizes, int n_in,
                              void* d_out, int out_size, void* d_ws, size_t ws_size,
                              hipStream_t stream) {
  const float* x       = (const float*)d_in[0];
  const float* centers = (const float*)d_in[1];
  const float* sigmas  = (const float*)d_in[2];
  const float* w_lin   = (const float*)d_in[3];
  const float* b_lin   = (const float*)d_in[4];
  float* out = (float*)d_out;

  char* ws = (char*)d_ws;
  size_t aPlane = (size_t)N_INPUT * KTOT * sizeof(__bf16);  // 16 MB
  size_t bPlane = (size_t)NUM_C   * KTOT * sizeof(__bf16);  //  4 MB
  __bf16* AH = (__bf16*)(ws);
  __bf16* AL = (__bf16*)(ws + aPlane);
  __bf16* BH = (__bf16*)(ws + 2 * aPlane);
  __bf16* BL = (__bf16*)(ws + 2 * aPlane + bPlane);
  float*  c2 = (float*)(ws + 2 * aPlane + 2 * bPlane);
  float*  partials = (float*)(ws + 2 * aPlane + 2 * bPlane + (size_t)NUM_C * sizeof(float));

  rbf_prep_a<<<N_INPUT * DIM / 256, 256, 0, stream>>>(x, AH, AL);
  rbf_prep_b<<<NUM_C * DIM / 256, 256, 0, stream>>>(centers, sigmas, BH, BL);
  rbf_prep_c2<<<NUM_C, 256, 0, stream>>>(centers, sigmas, c2);
  dim3 grid(N_INPUT / TILE_M, NUM_C / TILE_N);
  rbf_main<<<grid, 256, 0, stream>>>(AH, AL, BH, BL, c2, w_lin, partials);
  rbf_finish<<<N_INPUT / 256, 256, 0, stream>>>(partials, b_lin, out);
}
